// CausalSelfAttentionV1_74096775791136
// MI455X (gfx1250) — matrix operand order repaired and checked by reading, not compiled, not run on hardware
//
#include <hip/hip_runtime.h>
#include <stdint.h>

typedef __attribute__((ext_vector_type(16))) __bf16 v16bf;
typedef __attribute__((ext_vector_type(8)))  float  v8f;

union ABFrag { v16bf v; int4 q[2]; };

// A-fragment (16x32 bf16): rowbase points at this lane's M-row (32 contiguous K elems).
// lanes 0-15: K 0..7 and 16..23 ; lanes 16-31: K 8..15 and 24..31
__device__ inline v16bf load_afrag(const __bf16* rowbase, int lane) {
  ABFrag u;
  const int off = (lane < 16) ? 0 : 8;
  u.q[0] = *(const int4*)(rowbase + off);
  u.q[1] = *(const int4*)(rowbase + off + 16);
  return u.v;
}

// B-fragment (32x16 bf16): column n = lane&15, stride `stride` elems between columns,
// K contiguous within a column. lanes 0-15: K 0..7 and 16..23 ; lanes 16-31: K 8..15 and 24..31
__device__ inline v16bf load_bfrag(const __bf16* base, int stride, int lane) {
  const __bf16* p = base + (size_t)(lane & 15) * stride + ((lane < 16) ? 0 : 8);
  ABFrag u;
  u.q[0] = *(const int4*)p;
  u.q[1] = *(const int4*)(p + 16);
  return u.v;
}

__device__ inline v8f wmma_bf16(v16bf a, v16bf b, v8f c) {
  return __builtin_amdgcn_wmma_f32_16x16x32_bf16(false, a, false, b, (short)0, c,
                                                 false, false);
}

// Async 16-byte copy global -> LDS (bypasses VGPRs, tracked by ASYNCcnt).
__device__ inline void async_copy16(uint32_t ldsoff, const __bf16* g) {
  asm volatile("global_load_async_to_lds_b128 %0, %1, off"
               :: "v"(ldsoff), "v"(g)
               : "memory");
}

__device__ inline void wait_asynccnt0() {
  asm volatile("s_wait_asynccnt 0x0" ::: "memory");
}

// ---------------- LayerNorm (f32 in, bf16 out), one token per 256-thread block
__global__ __launch_bounds__(256) void ln_kernel(const float* __restrict__ x,
                                                 const float* __restrict__ gamma,
                                                 const float* __restrict__ beta,
                                                 __bf16* __restrict__ xn) {
  const int row = blockIdx.x;
  const int tid = threadIdx.x;
  const float4 v = ((const float4*)(x + (size_t)row * 1024))[tid];
  float s  = v.x + v.y + v.z + v.w;
  float s2 = v.x * v.x + v.y * v.y + v.z * v.z + v.w * v.w;
#pragma unroll
  for (int off = 16; off > 0; off >>= 1) {
    s  += __shfl_xor(s,  off, 32);
    s2 += __shfl_xor(s2, off, 32);
  }
  __shared__ float rs[8], rs2[8];
  if ((tid & 31) == 0) { rs[tid >> 5] = s; rs2[tid >> 5] = s2; }
  __syncthreads();
  float ts = 0.f, ts2 = 0.f;
#pragma unroll
  for (int j = 0; j < 8; ++j) { ts += rs[j]; ts2 += rs2[j]; }
  const float mu  = ts * (1.0f / 1024.0f);
  const float var = ts2 * (1.0f / 1024.0f) - mu * mu;
  const float inv = rsqrtf(var + 1e-5f);
  const int c = tid * 4;
  const float4 g  = ((const float4*)gamma)[tid];
  const float4 be = ((const float4*)beta)[tid];
  __bf16* o = xn + (size_t)row * 1024 + c;
  o[0] = (__bf16)((v.x - mu) * inv * g.x + be.x);
  o[1] = (__bf16)((v.y - mu) * inv * g.y + be.y);
  o[2] = (__bf16)((v.z - mu) * inv * g.z + be.z);
  o[3] = (__bf16)((v.w - mu) * inv * g.w + be.w);
}

// ---------------- f32 -> bf16 convert
__global__ void cvt_bf16_kernel(const float* __restrict__ in, __bf16* __restrict__ out,
                                int n) {
  const int i = blockIdx.x * 256 + threadIdx.x;
  if (i < n) out[i] = (__bf16)in[i];
}

// ---------------- Tiled bf16 GEMM: C[M,N] = A[M,K] @ B[K,N] + bias
// Block tile 128x128, 8 waves (4x2), wave tile 32x64 (2x4 wmma), K-step 32.
// A staged via async global->LDS copies, double-buffered; B transposed via VALU.
// MODE 0: QKV epilogue (scatter to Q [b,h,t,d] scaled, K [b,h,t,d], V^T [b,h,d,t])
// MODE 1: f32 output to outF.
template <int MODE>
__global__ __launch_bounds__(256) void gemm_bf16_kernel(
    const __bf16* __restrict__ A, const __bf16* __restrict__ Bm,
    const float* __restrict__ bias, float* __restrict__ outF,
    __bf16* __restrict__ Qb, __bf16* __restrict__ Kb, __bf16* __restrict__ Vt,
    int M, int N, int K) {
  __shared__ __bf16 As[2][128 * 32];   // row-major [m][k], double-buffered
  __shared__ __bf16 Bts[2][128 * 32];  // transposed [n][k], double-buffered
  const int tid   = threadIdx.x;
  const int lane  = tid & 31, wave = tid >> 5;
  const int waveM = wave >> 1, waveN = wave & 1;
  const int m0 = blockIdx.y * 128, n0 = blockIdx.x * 128;
  v8f acc[2][4] = {};
  const int arow = tid >> 1, ah = (tid & 1) * 16;
  const int bk = tid >> 3, bc = (tid & 7) * 16;

  auto stageA = [&](int buf, int k0) {
    const __bf16* asrc = A + (size_t)(m0 + arow) * K + k0 + ah;
    const uint32_t dst =
        (uint32_t)(uintptr_t)(&As[buf][arow * 32 + ah]);
    async_copy16(dst, asrc);
    async_copy16(dst + 16, asrc + 8);
  };
  auto stageB = [&](int buf, int k0) {
    const __bf16* bsrc = Bm + (size_t)(k0 + bk) * N + n0 + bc;
    __bf16 tmp[16];
    *(int4*)tmp       = *(const int4*)bsrc;
    *(int4*)(tmp + 8) = *(const int4*)(bsrc + 8);
#pragma unroll
    for (int j = 0; j < 16; ++j) Bts[buf][(bc + j) * 32 + bk] = tmp[j];
  };

  stageA(0, 0);
  stageB(0, 0);
  wait_asynccnt0();
  __syncthreads();

  const int nIter = K >> 5;
  for (int it = 0; it < nIter; ++it) {
    const int buf = it & 1;
    if (it + 1 < nIter) {
      stageA(buf ^ 1, (it + 1) * 32);
      __builtin_prefetch(Bm + (size_t)((it + 2) * 32 + bk) * N + n0 + bc, 0, 1);
      stageB(buf ^ 1, (it + 1) * 32);
    }
    v16bf af[2], bfr[4];
#pragma unroll
    for (int sm = 0; sm < 2; ++sm)
      af[sm] = load_afrag(
          &As[buf][(size_t)(waveM * 32 + sm * 16 + (lane & 15)) * 32], lane);
#pragma unroll
    for (int sn = 0; sn < 4; ++sn)
      bfr[sn] =
          load_bfrag(&Bts[buf][(size_t)(waveN * 64 + sn * 16) * 32], 32, lane);
#pragma unroll
    for (int sm = 0; sm < 2; ++sm)
#pragma unroll
      for (int sn = 0; sn < 4; ++sn)
        acc[sm][sn] = wmma_bf16(af[sm], bfr[sn], acc[sm][sn]);
    wait_asynccnt0();
    __syncthreads();
  }
  // Epilogue. C/D layout: lane 0-15 -> N=lane, M=vgpr i; lane 16-31 -> M = i+8.
#pragma unroll
  for (int sm = 0; sm < 2; ++sm)
#pragma unroll
    for (int sn = 0; sn < 4; ++sn)
#pragma unroll
      for (int i = 0; i < 8; ++i) {
        float v = acc[sm][sn][i];
        const int m = m0 + waveM * 32 + sm * 16 + ((lane < 16) ? i : i + 8);
        const int c = n0 + waveN * 64 + sn * 16 + (lane & 15);
        v += bias[c];
        if (MODE == 1) {
          outF[(size_t)m * N + c] = v;
        } else {
          const int bIdx = m >> 10, t = m & 1023;
          if (c < 1024) {
            const int h = c >> 6, d = c & 63;
            Qb[(((size_t)(bIdx * 16 + h)) * 1024 + t) * 64 + d] =
                (__bf16)(v * 0.125f);  // pre-scale by 1/sqrt(64)
          } else if (c < 2048) {
            const int c2 = c - 1024, h = c2 >> 6, d = c2 & 63;
            Kb[(((size_t)(bIdx * 16 + h)) * 1024 + t) * 64 + d] = (__bf16)v;
          } else {
            const int c2 = c - 2048, h = c2 >> 6, d = c2 & 63;
            Vt[(((size_t)(bIdx * 16 + h)) * 64 + d) * 1024 + t] = (__bf16)v;
          }
        }
      }
}

// ---------------- Flash attention: one wave = 16 query rows, key blocks of 32.
__global__ __launch_bounds__(128) void attn_kernel(
    const __bf16* __restrict__ Qb, const __bf16* __restrict__ Kb,
    const __bf16* __restrict__ Vt, const int* __restrict__ x_lens,
    __bf16* __restrict__ attout) {
  __shared__ __bf16 pbuf[4][16 * 32];  // per-wave P relayout buffer
  const int lane = threadIdx.x & 31;
  const int wave = threadIdx.x >> 5;
  const int b = blockIdx.z, h = blockIdx.y;
  const int tile = blockIdx.x * 4 + wave;
  const int t_lo = tile * 16;
  const int xlen = x_lens[b];
  const __bf16* Qh = Qb + ((size_t)(b * 16 + h) * 1024) * 64;
  const __bf16* Kh = Kb + ((size_t)(b * 16 + h) * 1024) * 64;
  const __bf16* Vh = Vt + ((size_t)(b * 16 + h) * 64) * 1024;
  __bf16* pb = pbuf[wave];

  const v16bf qa0 = load_afrag(Qh + (size_t)(t_lo + (lane & 15)) * 64, lane);
  const v16bf qa1 = load_afrag(Qh + (size_t)(t_lo + (lane & 15)) * 64 + 32, lane);

  v8f o[4] = {};
  float mrow[8], lrow[8];
#pragma unroll
  for (int i = 0; i < 8; ++i) { mrow[i] = -3.0e38f; lrow[i] = 0.0f; }

  const int tmax = t_lo + 15;
  const int send = (tmax < xlen - 1) ? tmax : (xlen - 1);
  const int rloc = (lane < 16) ? 0 : 8;
  for (int s0 = 0; s0 <= send; s0 += 32) {
    // S = Q @ K^T for 32 keys (two 16-wide score tiles), K-dim 64 split 32+32
    v8f sv0 = {}, sv1 = {};
    sv0 = wmma_bf16(qa0, load_bfrag(Kh + (size_t)s0 * 64, 64, lane), sv0);
    sv0 = wmma_bf16(qa1, load_bfrag(Kh + (size_t)s0 * 64 + 32, 64, lane), sv0);
    sv1 = wmma_bf16(qa0, load_bfrag(Kh + (size_t)(s0 + 16) * 64, 64, lane), sv1);
    sv1 = wmma_bf16(qa1, load_bfrag(Kh + (size_t)(s0 + 16) * 64 + 32, 64, lane), sv1);

    const int key0 = s0 + (lane & 15);
    const int key1 = key0 + 16;
#pragma unroll
    for (int i = 0; i < 8; ++i) {
      const int r = t_lo + rloc + i;
      float a  = sv0[i]; if (key0 > r || key0 >= xlen) a  = -1.0e9f;
      float bb = sv1[i]; if (key1 > r || key1 >= xlen) bb = -1.0e9f;
      float mx = fmaxf(a, bb);
#pragma unroll
      for (int ox = 1; ox < 16; ox <<= 1) mx = fmaxf(mx, __shfl_xor(mx, ox, 32));
      const float mnew  = fmaxf(mrow[i], mx);
      const float alpha = __expf(mrow[i] - mnew);
      const float p0 = __expf(a - mnew);
      const float p1 = __expf(bb - mnew);
      float rsum = p0 + p1;
#pragma unroll
      for (int ox = 1; ox < 16; ox <<= 1) rsum += __shfl_xor(rsum, ox, 32);
      lrow[i] = lrow[i] * alpha + rsum;
      mrow[i] = mnew;
#pragma unroll
      for (int dt = 0; dt < 4; ++dt) o[dt][i] *= alpha;
      // stash P row-major in LDS for A-fragment relayout (same-wave DS is in-order)
      pb[(rloc + i) * 32 + (lane & 15)]      = (__bf16)p0;
      pb[(rloc + i) * 32 + 16 + (lane & 15)] = (__bf16)p1;
    }
    const v16bf pf = load_afrag(pb + (size_t)(lane & 15) * 32, lane);
#pragma unroll
    for (int dt = 0; dt < 4; ++dt) {
      const v16bf vb = load_bfrag(Vh + (size_t)(dt * 16) * 1024 + s0, 1024, lane);
      o[dt] = wmma_bf16(pf, vb, o[dt]);
    }
  }
#pragma unroll
  for (int i = 0; i < 8; ++i) {
    const float inv = 1.0f / lrow[i];
    const int t = t_lo + rloc + i;
#pragma unroll
    for (int dt = 0; dt < 4; ++dt) {
      const int col = h * 64 + dt * 16 + (lane & 15);
      attout[((size_t)(b * 1024 + t)) * 1024 + col] = (__bf16)(o[dt][i] * inv);
    }
  }
}

extern "C" void kernel_launch(void* const* d_in, const int* in_sizes, int n_in,
                              void* d_out, int out_size, void* d_ws, size_t ws_size,
                              hipStream_t stream) {
  (void)in_sizes; (void)n_in; (void)out_size; (void)ws_size;
  const float* x        = (const float*)d_in[0];
  const int*   x_lens   = (const int*)d_in[1];
  const float* ln_gamma = (const float*)d_in[2];
  const float* ln_beta  = (const float*)d_in[3];
  const float* w_qkv    = (const float*)d_in[4];
  const float* b_qkv    = (const float*)d_in[5];
  const float* w_out    = (const float*)d_in[6];
  const float* b_out    = (const float*)d_in[7];
  float* out = (float*)d_out;

  char* w = (char*)d_ws;
  auto take = [&](size_t bytes) {
    char* p = w;
    w += (bytes + 255) & ~(size_t)255;
    return p;
  };
  __bf16* xn     = (__bf16*)take((size_t)8192 * 1024 * 2);
  __bf16* wqkv   = (__bf16*)take((size_t)1024 * 3072 * 2);
  __bf16* wout   = (__bf16*)take((size_t)1024 * 1024 * 2);
  __bf16* Qb     = (__bf16*)take((size_t)8 * 16 * 1024 * 64 * 2);
  __bf16* Kb     = (__bf16*)take((size_t)8 * 16 * 1024 * 64 * 2);
  __bf16* Vt     = (__bf16*)take((size_t)8 * 16 * 1024 * 64 * 2);
  __bf16* attout = (__bf16*)take((size_t)8192 * 1024 * 2);

  ln_kernel<<<8192, 256, 0, stream>>>(x, ln_gamma, ln_beta, xn);
  cvt_bf16_kernel<<<(1024 * 3072 + 255) / 256, 256, 0, stream>>>(w_qkv, wqkv,
                                                                 1024 * 3072);
  cvt_bf16_kernel<<<(1024 * 1024 + 255) / 256, 256, 0, stream>>>(w_out, wout,
                                                                 1024 * 1024);
  gemm_bf16_kernel<0><<<dim3(3072 / 128, 8192 / 128), 256, 0, stream>>>(
      xn, wqkv, b_qkv, nullptr, Qb, Kb, Vt, 8192, 3072, 1024);
  attn_kernel<<<dim3(16, 16, 8), 128, 0, stream>>>(Qb, Kb, Vt, x_lens, attout);
  gemm_bf16_kernel<1><<<dim3(1024 / 128, 8192 / 128), 256, 0, stream>>>(
      attout, wout, b_out, out, nullptr, nullptr, nullptr, 8192, 1024, 1024);
}
